// EagerAttention_56143812493837
// MI455X (gfx1250) — compile-verified, matrix operand order checked by reading, not run on hardware
//
#include <hip/hip_runtime.h>
#include <cstdint>
#include <cstddef>

// ---------------------------------------------------------------------------
// EagerAttention for MI455X (gfx1250, wave32, WMMA + TDM).
// Pipeline: f32->bf16 convert, QKV GEMMs (bf16 WMMA, f32 acc), RoPE + pack,
// flash-attention (TDM-staged K/V tiles in LDS, online softmax, WMMA QK^T
// and PV), output GEMM.  Workspace use: ~113 MB.
// ---------------------------------------------------------------------------

typedef __attribute__((ext_vector_type(16))) __bf16 bf16x16;
typedef __attribute__((ext_vector_type(8)))  __bf16 bf16x8;
typedef __attribute__((ext_vector_type(4)))  __bf16 bf16x4;
typedef __attribute__((ext_vector_type(8)))  float  f32x8;
typedef __attribute__((ext_vector_type(4)))  unsigned int u32x4;
typedef __attribute__((ext_vector_type(8)))  int    i32x8;
typedef __attribute__((ext_vector_type(4)))  int    i32x4;

constexpr int BB  = 2;
constexpr int TT  = 2048;
constexpr int DM  = 2048;
constexpr int NH  = 16;
constexpr int KVH = 4;
constexpr int HD  = 128;
constexpr int BT  = BB * TT;            // 4096 token rows
constexpr float QK_SCALE  = 0.08838834764831845f;   // 1/sqrt(128)
constexpr float NEG_BIG   = -3.0e38f;
constexpr float LOG_THETA = 13.815510557964274f;    // ln(1e6)

#define DEVFN __device__ __forceinline__

// Load one 16-bit WMMA operand fragment covering K = [0,32) of a row-major
// row.  Per the CDNA5 16-bit A/B layout, lanes 0-15 hold K {0..7,16..23} and
// lanes 16-31 hold K {8..15,24..31} of row (lane&15).  rowp must point at the
// row's K-base for this fragment.  Works for global or LDS-backed pointers.
DEVFN bf16x16 ld_frag32(const __bf16* rowp, int lh) {
  const int off = lh * 8;
  bf16x8 lo = *reinterpret_cast<const bf16x8*>(rowp + off);
  bf16x8 hi = *reinterpret_cast<const bf16x8*>(rowp + off + 16);
  bf16x16 r;
#pragma unroll
  for (int i = 0; i < 8; ++i) { r[i] = lo[i]; r[i + 8] = hi[i]; }
  return r;
}

DEVFN f32x8 wmma_bf16(bf16x16 a, bf16x16 b, f32x8 c) {
  return __builtin_amdgcn_wmma_f32_16x16x32_bf16(
      /*neg_a=*/false, a, /*neg_b=*/false, b,
      /*c_mod=*/(short)0, c, /*reuse_a=*/false, /*reuse_b=*/false);
}

// ---------------------------------------------------------------------------
// TDM: async-load one 2D bf16 tile (tile_d1 rows x tile_d0 elems) from a
// row-major tensor (row stride = stride0 elements) into LDS at lds_off.
// D# layout per CDNA5 ISA ch.8 (group0: control/lds/global addr, group1:
// sizes/strides; groups 2-3 zero for 2D tiles).  data_size=1 -> 2-byte elems.
// This toolchain exposes the 6-arg builtin (g0, g1, g2, g3, extra, cpol).
// ---------------------------------------------------------------------------
DEVFN void tdm_load_2d_bf16(unsigned lds_off, const __bf16* gptr,
                            unsigned tile_d0, unsigned tile_d1,
                            unsigned tensor_d0, unsigned tensor_d1,
                            unsigned long stride0) {
  unsigned long ga = (unsigned long)(uintptr_t)gptr;
  u32x4 g0;
  g0[0] = 1u;                                         // count=1, user mode
  g0[1] = lds_off;                                    // lds_addr (bytes)
  g0[2] = (unsigned)(ga & 0xFFFFFFFFu);               // global_addr[31:0]
  g0[3] = (unsigned)((ga >> 32) & 0x01FFFFFFu)        // global_addr[56:32]
        | (2u << 30);                                 // type=2 ("image")
  i32x8 g1;
  g1[0] = (int)(1u << 16);                            // data_size=1 (2B)
  g1[1] = (int)((tensor_d0 & 0xFFFFu) << 16);         // tensor_dim0[15:0]
  g1[2] = (int)((tensor_d0 >> 16) | ((tensor_d1 & 0xFFFFu) << 16));
  g1[3] = (int)((tensor_d1 >> 16) | (tile_d0 << 16)); // tile_dim0
  g1[4] = (int)(tile_d1 & 0xFFFFu);                   // tile_dim1 (dim2=0)
  g1[5] = (int)(unsigned)(stride0 & 0xFFFFFFFFu);     // dim0_stride[31:0]
  g1[6] = (int)(unsigned)((stride0 >> 32) & 0xFFFFu); // dim0_stride[47:32]
  g1[7] = 0;
  i32x4 z4 = {};
  i32x8 z8 = {};
  __builtin_amdgcn_tensor_load_to_lds(g0, g1, z4, z4, z8, 0);
}

// ---------------------------------------------------------------------------
// f32 -> bf16 bulk convert (vectorized x4)
// ---------------------------------------------------------------------------
__global__ void cvt_f32_to_bf16(const float* __restrict__ in,
                                __bf16* __restrict__ out, int n4) {
  int i = blockIdx.x * blockDim.x + threadIdx.x;
  if (i >= n4) return;
  float4 v = reinterpret_cast<const float4*>(in)[i];
  bf16x4 o;
  o[0] = (__bf16)v.x; o[1] = (__bf16)v.y; o[2] = (__bf16)v.z; o[3] = (__bf16)v.w;
  reinterpret_cast<bf16x4*>(out)[i] = o;
}

// ---------------------------------------------------------------------------
// NT GEMM:  C[M,N] (f32) = A[M,K] (bf16, row-major) x B[N,K]^T (bf16, row-major)
// One wave computes a 64x64 C tile: 4x4 grid of 16x16 WMMA accumulators.
// ---------------------------------------------------------------------------
__global__ __launch_bounds__(256) void gemm_nt_bf16_f32(
    const __bf16* __restrict__ A, const __bf16* __restrict__ B,
    float* __restrict__ C, int M, int N, int K) {
  const int lane = threadIdx.x & 31;
  const int wave = threadIdx.x >> 5;
  const int wpb  = blockDim.x >> 5;
  const int wid  = blockIdx.x * wpb + wave;
  const int ntiles = N >> 6;
  const int wm = (wid / ntiles) << 6;
  const int wn = (wid % ntiles) << 6;
  if (wm >= M) return;
  const int r  = lane & 15;
  const int lh = lane >> 4;

  f32x8 acc[4][4] = {};

  for (int k0 = 0; k0 < K; k0 += 32) {
    if (k0 + 32 < K) {
      __builtin_prefetch(A + (size_t)(wm + r) * K + k0 + 32, 0, 1);
      __builtin_prefetch(B + (size_t)(wn + r) * K + k0 + 32, 0, 1);
    }
    bf16x16 af[4], bf[4];
#pragma unroll
    for (int i = 0; i < 4; ++i)
      af[i] = ld_frag32(A + (size_t)(wm + i * 16 + r) * K + k0, lh);
#pragma unroll
    for (int j = 0; j < 4; ++j)
      bf[j] = ld_frag32(B + (size_t)(wn + j * 16 + r) * K + k0, lh);
#pragma unroll
    for (int i = 0; i < 4; ++i)
#pragma unroll
      for (int j = 0; j < 4; ++j)
        acc[i][j] = wmma_bf16(af[i], bf[j], acc[i][j]);
  }

  // C-layout: element v of each accumulator is row (lh*8+v), column r.
#pragma unroll
  for (int i = 0; i < 4; ++i)
#pragma unroll
    for (int j = 0; j < 4; ++j)
#pragma unroll
      for (int v = 0; v < 8; ++v) {
        int m = wm + i * 16 + lh * 8 + v;
        int n = wn + j * 16 + r;
        C[(size_t)m * N + n] = acc[i][j][v];
      }
}

// ---------------------------------------------------------------------------
// RoPE (HF/llama convention) + pack f32 [B*T, H*128] -> bf16 [B,H,T,128].
// One thread per (bt, h, i) pair, i in [0,64): handles elements i and i+64.
// ---------------------------------------------------------------------------
__global__ void rope_pack(const float* __restrict__ src,
                          __bf16* __restrict__ dst, int H) {
  long idx = (long)blockIdx.x * blockDim.x + threadIdx.x;
  long total = (long)BT * H * 64;
  if (idx >= total) return;
  int  i  = (int)(idx & 63);
  long t2 = idx >> 6;
  int  h  = (int)(t2 % H);
  long bt = t2 / H;
  int  t  = (int)(bt % TT);
  int  b  = (int)(bt / TT);

  float inv = __expf(-(float)i * (2.0f / 128.0f) * LOG_THETA);
  float ang = (float)t * inv;
  float s, c;
  __sincosf(ang, &s, &c);

  const float* row = src + bt * (size_t)(H * HD) + (size_t)h * HD;
  float x1 = row[i];
  float x2 = row[i + 64];
  __bf16* drow = dst + ((((size_t)b * H + h) * TT) + t) * HD;
  drow[i]      = (__bf16)(x1 * c - x2 * s);
  drow[i + 64] = (__bf16)(x2 * c + x1 * s);
}

// ---------------------------------------------------------------------------
// Pack V: f32 [B*T, KVH*128] -> bf16 transposed [B, KVH, 128, T]
// (so the PV WMMA B-operand is contiguous along the kv (K) dimension).
// ---------------------------------------------------------------------------
__global__ void pack_vT(const float* __restrict__ v, __bf16* __restrict__ vT) {
  long idx = (long)blockIdx.x * blockDim.x + threadIdx.x;
  long total = (long)BT * (KVH * HD);
  if (idx >= total) return;
  int  col = (int)(idx & (KVH * HD - 1));
  long bt  = idx >> 9;                 // / 512
  int  t   = (int)(bt % TT);
  int  b   = (int)(bt / TT);
  int  kh  = col >> 7;
  int  dd  = col & 127;
  vT[((((size_t)b * KVH + kh) * HD) + dd) * TT + t] = (__bf16)v[idx];
}

// ---------------------------------------------------------------------------
// Flash attention: one wave per 16-row Q tile; 4 waves/block share one (b,h)
// (QT=128 is divisible by 4, so a block never straddles heads).  Wave 0
// TDM-loads the shared 32-key K tile (32x128) and V^T tile (128x32) into LDS;
// all waves consume them.  All waves run to the block's max causal bound --
// fully-masked subtiles are numerically inert in the online softmax.
// ---------------------------------------------------------------------------
__global__ __launch_bounds__(128) void attn_kernel(
    const __bf16* __restrict__ Q,   // [B,NH,T,128]
    const __bf16* __restrict__ Kc,  // [B,KVH,T,128]
    const __bf16* __restrict__ Vt,  // [B,KVH,128,T]
    __bf16* __restrict__ O) {       // [B*T, NH*128]
  __shared__ __align__(16) __bf16 kbuf[32][128];   // [kv][d]
  __shared__ __align__(16) __bf16 vbuf[128][32];   // [d][kv]
  __shared__ __align__(16) __bf16 pbuf[4][16][32]; // per-wave P repack

  const int lane = threadIdx.x & 31;
  const int wave = threadIdx.x >> 5;
  const int wid  = blockIdx.x * 4 + wave;
  const int QT   = TT / 16;
  const int q0   = (wid % QT) * 16;
  const int h    = (wid / QT) % NH;
  const int b    = wid / (QT * NH);
  const int kh   = h >> 2;            // GQA repeat_interleave(4)
  const int r    = lane & 15;
  const int lh   = lane >> 4;

  // Q fragments (A-operand), 4 k-chunks of 32 covering head_dim=128.
  const __bf16* qbase = Q + ((((size_t)b * NH + h) * TT) + (q0 + r)) * HD;
  bf16x16 qf[4];
#pragma unroll
  for (int c = 0; c < 4; ++c) qf[c] = ld_frag32(qbase + c * 32, lh);

  f32x8 o[8] = {};
  float mi[8], li[8];
#pragma unroll
  for (int v = 0; v < 8; ++v) { mi[v] = NEG_BIG; li[v] = 0.0f; }

  const __bf16* kbase = Kc + (((size_t)b * KVH + kh) * TT) * HD;
  const __bf16* vbase = Vt + (((size_t)b * KVH + kh) * HD) * TT;

  // Common trip count for the whole block (last wave's causal bound).
  const int q0_last = ((blockIdx.x * 4 + 3) % QT) * 16;
  const int kend    = q0_last + 16;

  for (int kt = 0; kt < kend; kt += 32) {
    // --- wave 0 stages the shared K / V^T tiles via the TDM ---
    if (wave == 0) {
      tdm_load_2d_bf16((unsigned)(uintptr_t)&kbuf[0][0],
                       kbase + (size_t)kt * HD,
                       /*tile_d0=*/HD, /*tile_d1=*/32,
                       /*tensor_d0=*/HD, /*tensor_d1=*/(unsigned)(TT - kt),
                       /*stride0=*/HD);
      tdm_load_2d_bf16((unsigned)(uintptr_t)&vbuf[0][0],
                       vbase + kt,
                       /*tile_d0=*/32, /*tile_d1=*/HD,
                       /*tensor_d0=*/(unsigned)(TT - kt), /*tensor_d1=*/HD,
                       /*stride0=*/TT);
      __builtin_amdgcn_s_wait_tensorcnt(0);
    }
    __syncthreads();

    // --- scores for two 16-key subtiles (operands from LDS) ---
    f32x8 s[2] = {};
#pragma unroll
    for (int sub = 0; sub < 2; ++sub) {
#pragma unroll
      for (int c = 0; c < 4; ++c) {
        bf16x16 kf = ld_frag32(&kbuf[sub * 16 + r][c * 32], lh);
        s[sub] = wmma_bf16(qf[c], kf, s[sub]);
      }
    }

    // --- scale + causal mask + row max (butterfly over 16-lane halves) ---
    float mt[8];
#pragma unroll
    for (int v = 0; v < 8; ++v) {
      int mrow = q0 + lh * 8 + v;
      float a0 = s[0][v] * QK_SCALE; if (kt + r      > mrow) a0 = NEG_BIG;
      float a1 = s[1][v] * QK_SCALE; if (kt + 16 + r > mrow) a1 = NEG_BIG;
      s[0][v] = a0; s[1][v] = a1;
      float t = fmaxf(a0, a1);
      t = fmaxf(t, __shfl_xor(t, 1));
      t = fmaxf(t, __shfl_xor(t, 2));
      t = fmaxf(t, __shfl_xor(t, 4));
      t = fmaxf(t, __shfl_xor(t, 8));
      mt[v] = t;
    }

    // --- online softmax update ---
    float alpha[8];
#pragma unroll
    for (int v = 0; v < 8; ++v) {
      float mnew = fmaxf(mi[v], mt[v]);
      alpha[v] = __expf(mi[v] - mnew);
      mi[v] = mnew;
      float p0 = __expf(s[0][v] - mnew);
      float p1 = __expf(s[1][v] - mnew);
      s[0][v] = p0; s[1][v] = p1;
      float t = p0 + p1;
      t += __shfl_xor(t, 1);
      t += __shfl_xor(t, 2);
      t += __shfl_xor(t, 4);
      t += __shfl_xor(t, 8);
      li[v] = li[v] * alpha[v] + t;
    }
#pragma unroll
    for (int j = 0; j < 8; ++j)
#pragma unroll
      for (int v = 0; v < 8; ++v) o[j][v] *= alpha[v];

    // --- repack P (C-layout) -> A-layout via per-wave LDS ---
#pragma unroll
    for (int v = 0; v < 8; ++v) {
      pbuf[wave][lh * 8 + v][r]      = (__bf16)s[0][v];
      pbuf[wave][lh * 8 + v][16 + r] = (__bf16)s[1][v];
    }
    asm volatile("s_wait_dscnt 0" ::: "memory");
    bf16x16 pf;
    {
      bf16x8 lo = *reinterpret_cast<const bf16x8*>(&pbuf[wave][r][lh * 8]);
      bf16x8 hi = *reinterpret_cast<const bf16x8*>(&pbuf[wave][r][lh * 8 + 16]);
#pragma unroll
      for (int i = 0; i < 8; ++i) { pf[i] = lo[i]; pf[i + 8] = hi[i]; }
    }

    // --- PV: 8 output chunks of 16 along head_dim (V from LDS) ---
#pragma unroll
    for (int j = 0; j < 8; ++j) {
      bf16x16 vfr = ld_frag32(&vbuf[j * 16 + r][0], lh);
      o[j] = wmma_bf16(pf, vfr, o[j]);
    }

    __syncthreads();   // protect kbuf/vbuf before next tile overwrites
  }

  // --- epilogue: normalize and store bf16 [bt, h*128 + d] ---
  __bf16* obase = O + ((size_t)(b * TT + q0)) * (NH * HD) + (size_t)h * HD;
#pragma unroll
  for (int v = 0; v < 8; ++v) {
    float inv = 1.0f / li[v];
    int m = lh * 8 + v;
#pragma unroll
    for (int j = 0; j < 8; ++j)
      obase[(size_t)m * (NH * HD) + j * 16 + r] = (__bf16)(o[j][v] * inv);
  }
}

// ---------------------------------------------------------------------------
// Host-side orchestration
// ---------------------------------------------------------------------------
extern "C" void kernel_launch(void* const* d_in, const int* in_sizes, int n_in,
                              void* d_out, int out_size, void* d_ws, size_t ws_size,
                              hipStream_t stream) {
  const float* x  = (const float*)d_in[0];
  const float* wq = (const float*)d_in[1];
  const float* wk = (const float*)d_in[2];
  const float* wv = (const float*)d_in[3];
  const float* wo = (const float*)d_in[4];
  float* out = (float*)d_out;

  const size_t XE  = (size_t)BT * DM;          // 8.39M
  const size_t WQE = (size_t)(NH * HD) * DM;   // 4.19M
  const size_t WKE = (size_t)(KVH * HD) * DM;  // 1.05M
  const size_t QE  = XE;
  const size_t KE  = (size_t)BT * (KVH * HD);  // 2.10M

  char* ws = (char*)d_ws;
  size_t off = 0;
  auto take = [&](size_t bytes) -> char* {
    char* p = ws + off;
    off += (bytes + 255) & ~(size_t)255;
    return p;
  };

  __bf16* xb  = (__bf16*)take(XE * 2);
  __bf16* wqb = (__bf16*)take(WQE * 2);
  __bf16* wkb = (__bf16*)take(WKE * 2);
  __bf16* wvb = (__bf16*)take(WKE * 2);
  __bf16* wob = (__bf16*)take(WQE * 2);
  __bf16* qpk = (__bf16*)take(QE * 2);   // RoPE'd Q  [B,NH,T,128]
  __bf16* kpk = (__bf16*)take(KE * 2);   // RoPE'd K  [B,KVH,T,128]
  __bf16* vtp = (__bf16*)take(KE * 2);   // V^T       [B,KVH,128,T]
  float*  qf  = (float*)take(QE * 4);
  float*  kf  = (float*)take(KE * 4);
  float*  vf  = (float*)take(KE * 4);
  __bf16* attb = xb;   // x_bf16 is dead after the QKV GEMMs

  const int thr = 256;
  auto cvt = [&](const float* src, __bf16* dst, size_t n) {
    int n4 = (int)(n / 4);
    cvt_f32_to_bf16<<<(n4 + thr - 1) / thr, thr, 0, stream>>>(src, dst, n4);
  };
  cvt(x,  xb,  XE);
  cvt(wq, wqb, WQE);
  cvt(wk, wkb, WKE);
  cvt(wv, wvb, WKE);
  cvt(wo, wob, WQE);

  auto gemm = [&](const __bf16* A, const __bf16* B, float* C, int M, int N, int K) {
    int waves  = (M / 64) * (N / 64);
    int blocks = (waves + 7) / 8;          // 8 waves / 256-thread block
    gemm_nt_bf16_f32<<<blocks, 256, 0, stream>>>(A, B, C, M, N, K);
  };
  gemm(xb, wqb, qf, BT, NH * HD, DM);      // Q = x @ w_q^T
  gemm(xb, wkb, kf, BT, KVH * HD, DM);     // K = x @ w_k^T
  gemm(xb, wvb, vf, BT, KVH * HD, DM);     // V = x @ w_v^T

  {
    long tq = (long)BT * NH * 64;
    rope_pack<<<(int)((tq + thr - 1) / thr), thr, 0, stream>>>(qf, qpk, NH);
    long tk = (long)BT * KVH * 64;
    rope_pack<<<(int)((tk + thr - 1) / thr), thr, 0, stream>>>(kf, kpk, KVH);
    long tv = (long)BT * (KVH * HD);
    pack_vT<<<(int)((tv + thr - 1) / thr), thr, 0, stream>>>(vf, vtp);
  }

  {
    int waves  = BB * NH * (TT / 16);      // 4096
    int blocks = waves / 4;                // 4 waves / 128-thread block
    attn_kernel<<<blocks, 128, 0, stream>>>(qpk, kpk, vtp, attb);
  }

  gemm(attb, wob, out, BT, DM, DM);        // out = attn @ w_o^T
}
